// D_Align_54786602828117
// MI455X (gfx1250) — matrix operand order repaired and checked by reading, not compiled, not run on hardware
//
#include <hip/hip_runtime.h>
#include <stdint.h>

#define Bb 4
#define Hh 128
#define Ww 128
// NF=64, G=8, CG=8, KK=9

typedef __attribute__((ext_vector_type(16))) __bf16 v16bf;
typedef __attribute__((ext_vector_type(8)))  float  v8f;

struct U8x32 { uint4 lo, hi; };

static __device__ __forceinline__ v16bf load_frag(const unsigned short* p) {
  U8x32 u;
  u.lo = *(const uint4*)(p);
  u.hi = *(const uint4*)(p + 8);
  return __builtin_bit_cast(v16bf, u);
}
static __device__ __forceinline__ v16bf load_frag2(const unsigned short* p0,
                                                   const unsigned short* p1) {
  U8x32 u;
  u.lo = *(const uint4*)(p0);
  u.hi = *(const uint4*)(p1);
  return __builtin_bit_cast(v16bf, u);
}

// f32 -> bf16 round-to-nearest-even
static __device__ __forceinline__ unsigned short f2bf(float f) {
  uint32_t u = __builtin_bit_cast(uint32_t, f);
  u += 0x7FFFu + ((u >> 16) & 1u);
  return (unsigned short)(u >> 16);
}

// CDNA5 async global->LDS copy (ASYNCcnt-tracked, ISA 10 / 15.18.3 op 98).
// lds_off: byte offset of destination within the workgroup LDS allocation.
static __device__ __forceinline__ void async_load_b128(unsigned lds_off, uint64_t gaddr) {
  asm volatile("global_load_async_to_lds_b128 %0, %1, off"
               :: "v"(lds_off), "v"(gaddr) : "memory");
}
static __device__ __forceinline__ void wait_async0() {
  asm volatile("s_wait_asynccnt 0" ::: "memory");
}

// A-fragment (16x32 bf16) per-lane K index for packed position p (0..15)
// contiguous order: K = 16*half + p (matches B fills)
static __device__ __forceinline__ int a_k_of(int half, int p) {
  return half * 16 + p;
}

// ---------------- weight repack kernels ----------------
// wp1: [q=4][kc=36][lane=32][16]  K-order: tap*128 + c  (tap = ky*3+kx)
__global__ void repack_w1(const float* __restrict__ w, unsigned short* __restrict__ out) {
  int e = blockIdx.x * 256 + threadIdx.x;
  if (e >= 4 * 36 * 512) return;
  int p = e & 15, lane = (e >> 4) & 31;
  int rest = e >> 9;
  int kc = rest % 36, q = rest / 36;
  int m = lane & 15, half = lane >> 4;
  int och = q * 16 + m;
  int tap = kc >> 2, cchunk = kc & 3;
  int c = cchunk * 32 + a_k_of(half, p);
  int ty = tap / 3, tx = tap % 3;
  out[e] = f2bf(w[((och * 128 + c) * 3 + ty) * 3 + tx]);
}

// wp2: [q=16 (216 padded)][kc=18][lane=32][16]  K-order: tap*64 + c
__global__ void repack_w2(const float* __restrict__ w, unsigned short* __restrict__ out) {
  int e = blockIdx.x * 256 + threadIdx.x;
  if (e >= 16 * 18 * 512) return;
  int p = e & 15, lane = (e >> 4) & 31;
  int rest = e >> 9;
  int kc = rest % 18, q = rest / 18;
  int m = lane & 15, half = lane >> 4;
  int och = q * 16 + m;
  int tap = kc >> 1, cchunk = kc & 1;
  int c = cchunk * 32 + a_k_of(half, p);
  int ty = tap / 3, tx = tap % 3;
  float v = (och < 216) ? w[((och * 64 + c) * 3 + ty) * 3 + tx] : 0.0f;
  out[e] = f2bf(v);
}

// wp3: [q=4][kc=18][lane=32][16]  K-order: (g*9+kk)*8 + cg
__global__ void repack_w3(const float* __restrict__ w, unsigned short* __restrict__ out) {
  int e = blockIdx.x * 256 + threadIdx.x;
  if (e >= 4 * 18 * 512) return;
  int p = e & 15, lane = (e >> 4) & 31;
  int rest = e >> 9;
  int kc = rest % 18, q = rest / 18;
  int m = lane & 15, half = lane >> 4;
  int och = q * 16 + m;
  int Kidx = kc * 32 + a_k_of(half, p);
  int t = Kidx >> 3, cg = Kidx & 7;
  int g = t / 9, kk = t % 9;
  int cin = g * 8 + cg;
  int ty = kk / 3, tx = kk % 3;
  out[e] = f2bf(w[((och * 64 + cin) * 3 + ty) * 3 + tx]);
}

// ---------------- conv1: concat(nbr,ref)[128] -> off_feat (NHWC bf16, LReLU) -------------
#define C1PAD 136  // LDS channel pitch (shorts): 272B = +4 banks per column, 16B aligned
__global__ __launch_bounds__(256, 2) void conv1_kernel(
    const float* __restrict__ nbr, const float* __restrict__ ref,
    const unsigned short* __restrict__ wp1, const float* __restrict__ bias,
    unsigned short* __restrict__ off_feat) {
  __shared__ __align__(16) unsigned short lx[3 * 34 * C1PAD];
  int strip = blockIdx.x;
  int xs = (strip & 3) * 32;
  int y = (strip >> 2) & 127;
  int b = strip >> 9;
  int tid = threadIdx.x;

  // coalesced staging: column fastest across threads
  for (int e = tid; e < 128 * 3 * 34; e += 256) {
    int col = e % 34;
    int tmp = e / 34;
    int row = tmp % 3;
    int c = tmp / 3;
    int gy = y + row - 1, gx = xs + col - 1;
    float v = 0.0f;
    if (gy >= 0 && gy < Hh && gx >= 0 && gx < Ww) {
      const float* src = (c < 64) ? nbr : ref;
      int cc = c & 63;
      v = src[(((size_t)b * 64 + cc) * Hh + gy) * Ww + gx];
    }
    lx[(row * 34 + col) * C1PAD + c] = f2bf(v);
  }
  __syncthreads();

  int lane = tid & 31, wave = tid >> 5;
  int wj = wave & 1, q = wave >> 1;          // 4 out-channel blocks x 2 pixel halves
  int laneN = lane & 15, half = lane >> 4;
  int px = wj * 16 + laneN;

  v8f acc = {};
  const unsigned short* wbase = wp1 + (size_t)q * 36 * 512;
  for (int kc = 0; kc < 36; ++kc) {
    int tap = kc >> 2, cchunk = kc & 3;
    int dy = tap / 3, dx = tap % 3;
    const unsigned short* bp = &lx[(dy * 34 + px + dx) * C1PAD + cchunk * 32 + half * 16];
    v16bf bf = load_frag(bp);
    v16bf af = load_frag(wbase + kc * 512 + lane * 16);
    acc = __builtin_amdgcn_wmma_f32_16x16x32_bf16(false, af, false, bf, (short)0, acc,
                                                  false, false);
  }
  // epilogue: 8 consecutive channels per lane -> one packed b128 store (NHWC)
  unsigned short h[8];
#pragma unroll
  for (int r2 = 0; r2 < 8; ++r2) {
    int och = q * 16 + 8 * half + r2;
    float v = acc[r2] + bias[och];
    v = (v >= 0.0f) ? v : 0.1f * v;
    h[r2] = f2bf(v);
  }
  uint4 pk;
  pk.x = (uint32_t)h[0] | ((uint32_t)h[1] << 16);
  pk.y = (uint32_t)h[2] | ((uint32_t)h[3] << 16);
  pk.z = (uint32_t)h[4] | ((uint32_t)h[5] << 16);
  pk.w = (uint32_t)h[6] | ((uint32_t)h[7] << 16);
  size_t obase = (((size_t)b * Hh + y) * Ww + xs + px) * 64 + q * 16 + half * 8;
  *(uint4*)&off_feat[obase] = pk;
}

// -------- fused conv2 + dcn: off_feat -> om (LDS only) -> sample -> einsum -> out --------
// LDS pool: [om_s: 216*32 f32 = 27648 B][ union{ lx: 102*64 bf16 = 13056 B ,
//                                                smp: 72*32*8 bf16 = 36864 B } ]
#define OMS_BYTES 27648
__global__ __launch_bounds__(256, 2) void conv2_dcn_kernel(
    const float* __restrict__ nbr, const unsigned short* __restrict__ off_feat,
    const unsigned short* __restrict__ wp2, const float* __restrict__ b_om,
    const unsigned short* __restrict__ wp3, const float* __restrict__ b_dcn,
    float* __restrict__ out) {
  __shared__ __align__(16) char pool[OMS_BYTES + 72 * 32 * 8 * 2];
  float* om_s = (float*)pool;                                  // [216][32]
  unsigned short* lx = (unsigned short*)(pool + OMS_BYTES);    // phase A: [102][64]
  unsigned short* smp = (unsigned short*)(pool + OMS_BYTES);   // phase B/C: [72][32][8]

  int strip = blockIdx.x;
  int xs = (strip & 3) * 32;
  int y = (strip >> 2) & 127;
  int b = strip >> 9;
  int tid = threadIdx.x;
  int lane = tid & 31, wave = tid >> 5;
  int laneN = lane & 15, half = lane >> 4;
  int wj = wave & 1;
  int px = wj * 16 + laneN;

  // ---- stage off_feat halo tile via async DMA; halo cells zero-filled ----
  for (int j = tid; j < 102 * 8; j += 256) {
    int part = j & 7;
    int cell = j >> 3;
    int col = cell % 34, row = cell / 34;
    int gy = y + row - 1, gx = xs + col - 1;
    int eoff = cell * 64 + part * 8;  // element offset in lx
    if (gy >= 0 && gy < Hh && gx >= 0 && gx < Ww) {
      const unsigned short* gp =
          off_feat + (((size_t)b * Hh + gy) * Ww + gx) * 64 + part * 8;
      async_load_b128((unsigned)(OMS_BYTES + eoff * 2), (uint64_t)(uintptr_t)gp);
    } else {
      *(uint4*)&lx[eoff] = make_uint4(0u, 0u, 0u, 0u);
    }
  }
  wait_async0();
  __syncthreads();

  // ---- phase A: conv2 WMMA, om kept in LDS ----
  {
    int qb = wave >> 1;  // blocks qb, qb+4, qb+8, qb+12 (Cout padded to 256)
    v8f a0 = {}, a1 = {}, a2 = {}, a3 = {};
    for (int kc = 0; kc < 18; ++kc) {
      int tap = kc >> 1, cchunk = kc & 1;
      int dy = tap / 3, dx = tap % 3;
      const unsigned short* bp = &lx[(dy * 34 + px + dx) * 64 + cchunk * 32 + half * 16];
      v16bf bf = load_frag(bp);
      v16bf f0 = load_frag(wp2 + ((size_t)(qb + 0) * 18 + kc) * 512 + lane * 16);
      v16bf f1 = load_frag(wp2 + ((size_t)(qb + 4) * 18 + kc) * 512 + lane * 16);
      v16bf f2 = load_frag(wp2 + ((size_t)(qb + 8) * 18 + kc) * 512 + lane * 16);
      v16bf f3 = load_frag(wp2 + ((size_t)(qb + 12) * 18 + kc) * 512 + lane * 16);
      a0 = __builtin_amdgcn_wmma_f32_16x16x32_bf16(false, f0, false, bf, (short)0, a0, false, false);
      a1 = __builtin_amdgcn_wmma_f32_16x16x32_bf16(false, f1, false, bf, (short)0, a1, false, false);
      a2 = __builtin_amdgcn_wmma_f32_16x16x32_bf16(false, f2, false, bf, (short)0, a2, false, false);
      a3 = __builtin_amdgcn_wmma_f32_16x16x32_bf16(false, f3, false, bf, (short)0, a3, false, false);
    }
#pragma unroll
    for (int t = 0; t < 4; ++t) {
      int q = qb + 4 * t;
      v8f acc = (t == 0) ? a0 : (t == 1) ? a1 : (t == 2) ? a2 : a3;
#pragma unroll
      for (int r2 = 0; r2 < 8; ++r2) {
        int och = q * 16 + 8 * half + r2;
        if (och < 216) om_s[och * 32 + px] = acc[r2] + b_om[och];
      }
    }
  }
  __syncthreads();  // lx dead from here; smp overlays it

  // ---- phase B: modulated bilinear samples into LDS ----
  for (int j = tid; j < 72 * 32; j += 256) {
    int p = j & 31;
    int t = j >> 5;  // t = g*9 + kk
    int g = t / 9, kk = t % 9;
    int x = xs + p;
    size_t hw = (size_t)Hh * Ww;
    float oy = om_s[t * 32 + p];
    float ox = om_s[(72 + t) * 32 + p];
    float mr = om_s[(144 + t) * 32 + p];
    float mask = 1.0f / (1.0f + __expf(-mr));
    float py = (float)y + (float)(kk / 3 - 1) + oy;
    float pxx = (float)x + (float)(kk % 3 - 1) + ox;
    float y0f = floorf(py), x0f = floorf(pxx);
    float wy = py - y0f, wx = pxx - x0f;
    int y0 = (int)y0f, x0 = (int)x0f;
    float accs[8];
#pragma unroll
    for (int c = 0; c < 8; ++c) accs[c] = 0.0f;
    const float* nb = nbr + ((size_t)b * 64 + g * 8) * hw;
#pragma unroll
    for (int corner = 0; corner < 4; ++corner) {
      int dyc = corner >> 1, dxc = corner & 1;
      int yy = y0 + dyc, xx = x0 + dxc;
      float wgt = (dyc ? wy : 1.0f - wy) * (dxc ? wx : 1.0f - wx);
      bool valid = (yy >= 0) && (yy < Hh) && (xx >= 0) && (xx < Ww);
      float wv = valid ? wgt : 0.0f;
      int yc = yy < 0 ? 0 : (yy > Hh - 1 ? Hh - 1 : yy);
      int xc = xx < 0 ? 0 : (xx > Ww - 1 ? Ww - 1 : xx);
      size_t idx = (size_t)yc * Ww + xc;
#pragma unroll
      for (int c = 0; c < 8; ++c) accs[c] += nb[c * hw + idx] * wv;
    }
    uint32_t pk0 = (uint32_t)f2bf(accs[0] * mask) | ((uint32_t)f2bf(accs[1] * mask) << 16);
    uint32_t pk1 = (uint32_t)f2bf(accs[2] * mask) | ((uint32_t)f2bf(accs[3] * mask) << 16);
    uint32_t pk2 = (uint32_t)f2bf(accs[4] * mask) | ((uint32_t)f2bf(accs[5] * mask) << 16);
    uint32_t pk3 = (uint32_t)f2bf(accs[6] * mask) | ((uint32_t)f2bf(accs[7] * mask) << 16);
    *(uint4*)&smp[(size_t)j * 8] = make_uint4(pk0, pk1, pk2, pk3);
  }
  __syncthreads();

  // ---- phase C: GEMM  out[64, px] = W[64, 576] x smp[576, px] ----
  {
    int q = wave >> 1;
    v8f acc = {};
    for (int kc = 0; kc < 18; ++kc) {
      int t0 = kc * 4 + half * 2;
      const unsigned short* b0 = &smp[((size_t)t0 * 32 + px) * 8];
      const unsigned short* b1 = &smp[((size_t)(t0 + 1) * 32 + px) * 8];
      v16bf bf = load_frag2(b0, b1);
      v16bf af = load_frag(wp3 + ((size_t)q * 18 + kc) * 512 + lane * 16);
      acc = __builtin_amdgcn_wmma_f32_16x16x32_bf16(false, af, false, bf, (short)0, acc,
                                                    false, false);
    }
#pragma unroll
    for (int r2 = 0; r2 < 8; ++r2) {
      int och = q * 16 + r2 + 8 * half;
      float v = acc[r2] + b_dcn[och];
      v = (v >= 0.0f) ? v : 0.1f * v;
      out[(((size_t)b * 64 + och) * Hh + y) * Ww + xs + px] = v;
    }
  }
}

extern "C" void kernel_launch(void* const* d_in, const int* in_sizes, int n_in,
                              void* d_out, int out_size, void* d_ws, size_t ws_size,
                              hipStream_t stream) {
  (void)in_sizes; (void)n_in; (void)out_size; (void)ws_size;
  const float* nbr    = (const float*)d_in[0];
  const float* ref    = (const float*)d_in[1];
  const float* w_off1 = (const float*)d_in[2];
  const float* b_off1 = (const float*)d_in[3];
  const float* w_om   = (const float*)d_in[4];
  const float* b_om   = (const float*)d_in[5];
  const float* w_dcn  = (const float*)d_in[6];
  const float* b_dcn  = (const float*)d_in[7];
  float* out = (float*)d_out;

  char* ws = (char*)d_ws;
  unsigned short* wp1      = (unsigned short*)(ws + 0);        // 147456 B
  unsigned short* wp2      = (unsigned short*)(ws + 147456);   // 294912 B
  unsigned short* wp3      = (unsigned short*)(ws + 442368);   // 73728 B
  unsigned short* off_feat = (unsigned short*)(ws + 516096);   // 8.39 MB (NHWC bf16)

  repack_w1<<<288, 256, 0, stream>>>(w_off1, wp1);
  repack_w2<<<576, 256, 0, stream>>>(w_om, wp2);
  repack_w3<<<144, 256, 0, stream>>>(w_dcn, wp3);

  int strips = Bb * Hh * (Ww / 32);  // 2048
  conv1_kernel<<<strips, 256, 0, stream>>>(nbr, ref, wp1, b_off1, off_feat);
  conv2_dcn_kernel<<<strips, 256, 0, stream>>>(nbr, off_feat, wp2, b_om, wp3, b_dcn, out);
}
